// PillarCrossAttention_85959475462563
// MI455X (gfx1250) — matrix operand order repaired and checked by reading, not compiled, not run on hardware
//
#include <hip/hip_runtime.h>
#include <math.h>

typedef __attribute__((ext_vector_type(16))) _Float16 v16h;
typedef __attribute__((ext_vector_type(8)))  float    v8f;

#define KW     3
#define CIN    64
#define HDIM   380
#define WDIM   380
#define BATCH  4
#define LQ     126              // HDIM / KW
#define NPIL   (LQ * LQ)        // pillars per (batch, shift)
#define NTOK   9                // KW*KW tokens per pillar
#define NWAVE  8
#define NTHR   (NWAVE * 32)

// K index inside a 16-bit A fragment (16x32), per CDNA5 ISA 7.12.2
__device__ __forceinline__ int a_k16(int e, int half) {
  int g = e >> 1, lo = e & 1;
  return ((g < 4) ? 0 : 16) + half * 8 + 2 * (g & 3) + lo;
}

__global__ __launch_bounds__(NTHR)
void pillar_attn_kernel(const float* __restrict__ x,
                        const float* __restrict__ Wq, const float* __restrict__ bq,
                        const float* __restrict__ Wk, const float* __restrict__ bk,
                        const float* __restrict__ Wv, const float* __restrict__ bv,
                        float* __restrict__ out)
{
  __shared__ float     peS[NTOK][CIN];           // positional encoding
  __shared__ _Float16  wS[3][CIN][CIN];          // [mat][out_chan][in_chan] f16
  __shared__ _Float16  qkvS[3][NWAVE][16][CIN];  // [mat][wave][token][chan]
  __shared__ _Float16  attS[NWAVE][16][16];      // attention probs

  const int tid = threadIdx.x;
  const int b   = blockIdx.z;
  const int si  = blockIdx.y / KW;
  const int sj  = blockIdx.y % KW;

  // ---- cooperative staging: PE table + f16 weights ----
  for (int idx = tid; idx < NTOK * CIN; idx += NTHR) {
    int p = idx / CIN, c = idx % CIN;
    float ang = (float)p * __powf(10000.0f, -(float)(c >> 1) * (1.0f / 16.0f));
    peS[p][c] = (c & 1) ? __cosf(ang) : __sinf(ang);
  }
  const float* Wsrc[3] = {Wq, Wk, Wv};
  for (int idx = tid; idx < 3 * CIN * CIN; idx += NTHR) {
    int mat = idx >> 12, r = idx & 4095;
    wS[mat][r >> 6][r & 63] = (_Float16)Wsrc[mat][r];
  }
  __syncthreads();

  const int wave = tid >> 5;
  const int lane = tid & 31;
  const int half = lane >> 4;
  const int l16  = lane & 15;

  const int pillar = blockIdx.x * NWAVE + wave;
  if (pillar >= NPIL) return;    // wave-uniform exit (no further barriers)

  const int py = pillar / LQ, px = pillar % LQ;
  const int h0 = si + py * KW, w0 = sj + px * KW;

  // ---- A fragments of (x + PE), tokens x 64 channels, f16 ----
  // rows 9..15 replicate token 0: k-cols masked in softmax, v-rows killed by att=0,
  // q-rows never stored, so padding is harmless.
  const int mtok = (l16 < NTOK) ? l16 : 0;
  const size_t HW = (size_t)HDIM * WDIM;
  const float* xb = x + (size_t)b * CIN * HW;
  const size_t pix = (size_t)(h0 + mtok / KW) * WDIM + (w0 + mtok % KW);
  v16h ax[2];
  #pragma unroll
  for (int kt = 0; kt < 2; ++kt) {
    #pragma unroll
    for (int e = 0; e < 16; ++e) {
      int c = kt * 32 + a_k16(e, half);
      ax[kt][e] = (_Float16)(xb[(size_t)c * HW + pix] + peS[mtok][c]);
    }
  }

  // ---- q,k,v = x @ W^T + b  (M=16 tokens, N=64, K=64) ----
  const float* bsrc[3] = {bq, bk, bv};
  #pragma unroll
  for (int mat = 0; mat < 3; ++mat) {
    #pragma unroll
    for (int n = 0; n < 4; ++n) {
      const int oc = n * 16 + l16;           // output channel this lane owns
      v8f acc = {};
      #pragma unroll
      for (int kt = 0; kt < 2; ++kt) {
        v16h bw;                             // B[K=in_chan][N=out_chan] = W[oc][ic]
        #pragma unroll
        for (int e = 0; e < 16; ++e)
          bw[e] = wS[mat][oc][kt * 32 + a_k16(e, half)];
        acc = __builtin_amdgcn_wmma_f32_16x16x32_f16(false, ax[kt], false, bw,
                                                     (short)0, acc, false, false);
      }
      const float bias = bsrc[mat][oc];
      #pragma unroll
      for (int g = 0; g < 8; ++g)
        qkvS[mat][wave][g + half * 8][oc] = (_Float16)(acc[g] + bias);
    }
  }

  // ---- scores = q @ k^T  (16x16, K=64 channels) ----
  v8f sc = {};
  #pragma unroll
  for (int kt = 0; kt < 2; ++kt) {
    v16h aq, bkf;
    #pragma unroll
    for (int e = 0; e < 16; ++e) {
      aq[e]  = qkvS[0][wave][l16][kt * 32 + a_k16(e, half)];   // A: row=token, K=chan
      bkf[e] = qkvS[1][wave][l16][kt * 32 + a_k16(e, half)];   // B[K=chan][N=token]=k[N][K]
    }
    sc = __builtin_amdgcn_wmma_f32_16x16x32_f16(false, aq, false, bkf,
                                                (short)0, sc, false, false);
  }

  // ---- masked softmax over key dim (one column per lane; valid cols < 9) ----
  #pragma unroll
  for (int g = 0; g < 8; ++g) {
    float s = (l16 < NTOK) ? sc[g] : -3.0e38f;
    float mx = s;
    #pragma unroll
    for (int off = 1; off < 16; off <<= 1)
      mx = fmaxf(mx, __shfl_xor(mx, off, 16));
    float e = (l16 < NTOK) ? __expf(s - mx) : 0.0f;
    float sum = e;
    #pragma unroll
    for (int off = 1; off < 16; off <<= 1)
      sum += __shfl_xor(sum, off, 16);
    sc[g] = e / sum;
  }

  // stash att (f16) and rebuild as A fragment (K = token dim, zero-padded to 32)
  #pragma unroll
  for (int g = 0; g < 8; ++g)
    attS[wave][g + half * 8][l16] = (_Float16)sc[g];

  v16h aat;
  #pragma unroll
  for (int e = 0; e < 16; ++e) {
    int kk = a_k16(e, half);
    aat[e] = (kk < 16) ? attS[wave][l16][kk] : (_Float16)0.0f;
  }

  // ---- o = att @ v ; atomic scatter-add into out channels 64..127 ----
  #pragma unroll
  for (int n = 0; n < 4; ++n) {
    v16h bvf;
    #pragma unroll
    for (int e = 0; e < 16; ++e) {
      int t = a_k16(e, half);                // key-token index
      bvf[e] = (t < 16) ? qkvS[2][wave][t][n * 16 + l16] : (_Float16)0.0f;
    }
    v8f o = {};
    o = __builtin_amdgcn_wmma_f32_16x16x32_f16(false, aat, false, bvf,
                                               (short)0, o, false, false);
    float* ob = out + ((size_t)b * 128 + 64 + n * 16 + l16) * HW;
    #pragma unroll
    for (int g = 0; g < 8; ++g) {
      int t = g + half * 8;                  // query token (row)
      if (t < NTOK)
        atomicAdd(&ob[(size_t)(h0 + t / KW) * WDIM + (w0 + t % KW)], o[g]);
    }
  }
}

// out[:, :64] = x ; out[:, 64:] = 0 (acc init before atomics)
__global__ void init_out_kernel(const float* __restrict__ x, float* __restrict__ out) {
  size_t i = (size_t)blockIdx.x * blockDim.x + threadIdx.x;
  const size_t HW = (size_t)HDIM * WDIM;
  const size_t tot = (size_t)BATCH * 128 * HW;
  if (i >= tot) return;
  size_t b = i / (128 * HW);
  size_t r = i % (128 * HW);
  size_t c = r / HW;
  size_t p = r % HW;
  out[i] = (c < 64) ? x[(b * 64 + c) * HW + p] : 0.0f;
}

// divide acc region by overlap-count mask
__global__ void finalize_kernel(float* __restrict__ out) {
  size_t i = (size_t)blockIdx.x * blockDim.x + threadIdx.x;
  const size_t HW = (size_t)HDIM * WDIM;
  const size_t tot = (size_t)BATCH * 64 * HW;
  if (i >= tot) return;
  size_t b = i / (64 * HW);
  size_t r = i % (64 * HW);
  size_t c = r / HW;
  size_t p = r % HW;
  int h = (int)(p / WDIM), w = (int)(p % WDIM);
  float ch = (float)(min(h, HDIM - KW) - max(0, h - KW + 1) + 1);
  float cw = (float)(min(w, WDIM - KW) - max(0, w - KW + 1) + 1);
  out[(b * 128 + 64 + c) * HW + p] /= (ch * cw);
}

extern "C" void kernel_launch(void* const* d_in, const int* in_sizes, int n_in,
                              void* d_out, int out_size, void* d_ws, size_t ws_size,
                              hipStream_t stream) {
  const float* x  = (const float*)d_in[0];
  const float* Wq = (const float*)d_in[1];
  const float* bq = (const float*)d_in[2];
  const float* Wk = (const float*)d_in[3];
  const float* bk = (const float*)d_in[4];
  const float* Wv = (const float*)d_in[5];
  const float* bv = (const float*)d_in[6];
  float* out = (float*)d_out;

  const size_t HW = (size_t)HDIM * WDIM;
  const size_t tot_out = (size_t)BATCH * 128 * HW;
  init_out_kernel<<<(unsigned)((tot_out + 255) / 256), 256, 0, stream>>>(x, out);

  dim3 grid((NPIL + NWAVE - 1) / NWAVE, KW * KW, BATCH);
  pillar_attn_kernel<<<grid, NTHR, 0, stream>>>(x, Wq, bq, Wk, bk, Wv, bv, out);

  const size_t tot_acc = (size_t)BATCH * 64 * HW;
  finalize_kernel<<<(unsigned)((tot_acc + 255) / 256), 256, 0, stream>>>(out);
}
